// LSTMCell_88828513616450
// MI455X (gfx1250) — matrix operand order repaired and checked by reading, not compiled, not run on hardware
//
#include <hip/hip_runtime.h>
#include <hip/hip_bf16.h>

#define T_STEPS 1024
#define BATCH   64
#define DIN     512
#define HDIM    512
#define NG      2048   // 4*H
#define KW      1024   // D+H (row length of transposed weights)
#define LN_EPS  1e-5f
#define NBLK    64     // persistent grid size
#define LDA     520    // padded LDS row stride (halves) for h tile

typedef __attribute__((ext_vector_type(16))) _Float16 v16h;
typedef __attribute__((ext_vector_type(8)))  _Float16 v8h;
typedef __attribute__((ext_vector_type(8)))  float    v8f;
typedef __attribute__((ext_vector_type(4)))  float    v4f;
typedef __attribute__((ext_vector_type(4)))  int      v4i;

#if defined(__HIP_DEVICE_COMPILE__) && \
    __has_builtin(__builtin_amdgcn_global_load_async_to_lds_b128) && \
    __has_builtin(__builtin_amdgcn_s_wait_asynccnt)
#define HAVE_ASYNC_LDS 1
typedef __attribute__((address_space(1))) v4i* gv4i_p;   // global int4*
typedef __attribute__((address_space(3))) v4i* lv4i_p;   // LDS int4*
#endif

// ---------------------------------------------------------------------------
// Wt[n][k] = (f16) W[k][n]   (W is [D+H, 4H] row-major f32)
// ---------------------------------------------------------------------------
__global__ __launch_bounds__(256) void prep_weights(const float* __restrict__ W,
                                                    _Float16* __restrict__ Wt) {
    int idx = blockIdx.x * 256 + threadIdx.x;   // 0 .. NG*KW-1
    int n = idx >> 10;
    int k = idx & (KW - 1);
    Wt[idx] = (_Float16)W[(long)k * NG + n];
}

__global__ void init_sync(unsigned* s) {
    if (threadIdx.x < 2) s[threadIdx.x] = 0u;
}

// ---------------------------------------------------------------------------
// Phase 1: pre[t*B+b][n] = x[t][b][:] . Wx[:][n]
// M = T*B = 65536, K = 512, N = 2048.  Block = 8 waves, block tile 64M x 128N,
// wave tile 16M x 64N (one A fragment shared by 4 WMMAs per K step).
// ---------------------------------------------------------------------------
__global__ __launch_bounds__(256) void pregate_gemm(const float* __restrict__ x,
                                                    const _Float16* __restrict__ Wt,
                                                    float* __restrict__ pre) {
    const int tid   = threadIdx.x;
    const int wv    = tid >> 5;
    const int lane  = tid & 31;
    const int lrow  = lane & 15;
    const int lhalf = lane >> 4;
    const int mBase = blockIdx.x * 64 + (wv >> 1) * 16;
    const int nBase = blockIdx.y * 128 + (wv & 1) * 64;

    const float* xrow = x + (long)(mBase + lrow) * DIN;

    v8f acc0 = {}, acc1 = {}, acc2 = {}, acc3 = {};
    for (int kk = 0; kk < DIN; kk += 32) {
        const v4f* q0 = (const v4f*)(xrow + kk + lhalf * 16);
        const v4f* q1 = (const v4f*)(xrow + kk + lhalf * 16 + 8);
        v4f f0 = q0[0], f1 = q0[1], f2 = q1[0], f3 = q1[1];
        v16h a;
        #pragma unroll
        for (int i = 0; i < 4; ++i) {
            a[i]      = (_Float16)f0[i];
            a[4 + i]  = (_Float16)f1[i];
            a[8 + i]  = (_Float16)f2[i];
            a[12 + i] = (_Float16)f3[i];
        }
        const _Float16* wb = Wt + kk + lhalf * 16;
        v16h b0 = *(const v16h*)(wb + (long)(nBase + 0  + lrow) * KW);
        v16h b1 = *(const v16h*)(wb + (long)(nBase + 16 + lrow) * KW);
        v16h b2 = *(const v16h*)(wb + (long)(nBase + 32 + lrow) * KW);
        v16h b3 = *(const v16h*)(wb + (long)(nBase + 48 + lrow) * KW);

        acc0 = __builtin_amdgcn_wmma_f32_16x16x32_f16(false, a, false, b0, (short)0, acc0, false, false);
        acc1 = __builtin_amdgcn_wmma_f32_16x16x32_f16(false, a, false, b1, (short)0, acc1, false, false);
        acc2 = __builtin_amdgcn_wmma_f32_16x16x32_f16(false, a, false, b2, (short)0, acc2, false, false);
        acc3 = __builtin_amdgcn_wmma_f32_16x16x32_f16(false, a, false, b3, (short)0, acc3, false, false);
    }
    float* prow = pre + (long)(mBase + lhalf * 8) * NG + nBase + lrow;
    #pragma unroll
    for (int r = 0; r < 8; ++r) {
        prow[(long)r * NG + 0]  = acc0[r];
        prow[(long)r * NG + 16] = acc1[r];
        prow[(long)r * NG + 32] = acc2[r];
        prow[(long)r * NG + 48] = acc3[r];
    }
}

// ---------------------------------------------------------------------------
// Device-wide sense barrier (persistent kernel, NBLK co-resident workgroups)
// ---------------------------------------------------------------------------
__device__ __forceinline__ void grid_sync(unsigned* cnt, unsigned* gen) {
    __syncthreads();
    if (threadIdx.x == 0) {
        unsigned g = __hip_atomic_load(gen, __ATOMIC_RELAXED, __HIP_MEMORY_SCOPE_AGENT);
        unsigned prev = __hip_atomic_fetch_add(cnt, 1u, __ATOMIC_ACQ_REL, __HIP_MEMORY_SCOPE_AGENT);
        if (prev == NBLK - 1u) {
            __hip_atomic_store(cnt, 0u, __ATOMIC_RELAXED, __HIP_MEMORY_SCOPE_AGENT);
            __hip_atomic_store(gen, g + 1u, __ATOMIC_RELEASE, __HIP_MEMORY_SCOPE_AGENT);
        } else {
            while (__hip_atomic_load(gen, __ATOMIC_ACQUIRE, __HIP_MEMORY_SCOPE_AGENT) == g)
                __builtin_amdgcn_s_sleep(2);
        }
    }
    __syncthreads();
}

__device__ __forceinline__ float sigmoidf_(float x) { return 1.f / (1.f + __expf(-x)); }

// ---------------------------------------------------------------------------
// Persistent recurrent kernel: 64 WGs x 256 threads, all 1024 timesteps.
// GEMM: WG g -> M-tile (g&3), N-block 128*(g>>2); wave tile 16x16, K=512,
//       weights register-resident (16 fragments), A staged in LDS (async path).
// Pointwise: WG g -> batch row g, cell state in registers.
// ---------------------------------------------------------------------------
__global__ __launch_bounds__(256) void lstm_persistent(
        const _Float16* __restrict__ Wt, const float* __restrict__ pre,
        const float* __restrict__ h0, const float* __restrict__ c0,
        const float* __restrict__ bias,
        const float* __restrict__ gg, const float* __restrict__ gb,
        const float* __restrict__ cg, const float* __restrict__ cb,
        _Float16* __restrict__ h_f16, float* __restrict__ gates,
        float* __restrict__ out, unsigned* __restrict__ sync) {
    __shared__ _Float16 hA[16 * LDA];
    __shared__ float gl[NG];
    __shared__ float red[16];

    const int g     = blockIdx.x;          // 0..63
    const int tid   = threadIdx.x;
    const int wv    = tid >> 5;
    const int lane  = tid & 31;
    const int lrow  = lane & 15;
    const int lhalf = lane >> 4;
    const int mBase = (g & 3) * 16;
    const int nBase = (g >> 2) * 128 + wv * 16;
    const int b     = g;                   // pointwise batch row

    // -------- register-resident recurrent weights: 16 frags = 128 VGPRs/lane
    v16h bw[16];
    {
        const _Float16* wb = Wt + DIN + lhalf * 16 + (long)(nBase + lrow) * KW;
        #pragma unroll
        for (int kk = 0; kk < 16; ++kk)
            bw[kk] = *(const v16h*)(wb + kk * 32);
    }

    // -------- initial state: c in registers, h seeded to global f16
    float creg[2];
    #pragma unroll
    for (int i = 0; i < 2; ++i) {
        int j = tid * 2 + i;
        creg[i] = c0[b * HDIM + j];
        h_f16[b * HDIM + j] = (_Float16)h0[b * HDIM + j];
    }
    grid_sync(sync, sync + 1);

    for (int t = 0; t < T_STEPS; ++t) {
        // ---- stage h rows [mBase, mBase+16) into LDS (16 KB, 1024 x 16B chunks)
        #pragma unroll
        for (int k = 0; k < 4; ++k) {
            int chunk = tid + k * 256;
            int row = chunk >> 6, col = chunk & 63;
            const _Float16* src = h_f16 + (mBase + row) * HDIM + col * 8;
            _Float16* dst = &hA[row * LDA + col * 8];
#ifdef HAVE_ASYNC_LDS
            __builtin_amdgcn_global_load_async_to_lds_b128(
                (gv4i_p)(uintptr_t)src,
                (lv4i_p)(unsigned)(uintptr_t)dst, 0, 0);
#else
            *(v8h*)dst = *(const v8h*)src;
#endif
        }
#ifdef HAVE_ASYNC_LDS
        __builtin_amdgcn_s_wait_asynccnt(0);
#endif
        __syncthreads();

        // ---- 16x16 GEMM tile, K = 512, A from LDS, B from registers
        v8f acc = {};
        #pragma unroll
        for (int kk = 0; kk < 16; ++kk) {
            v8h a0 = *(const v8h*)&hA[lrow * LDA + kk * 32 + lhalf * 16];
            v8h a1 = *(const v8h*)&hA[lrow * LDA + kk * 32 + lhalf * 16 + 8];
            v16h a;
            #pragma unroll
            for (int i = 0; i < 8; ++i) { a[i] = a0[i]; a[8 + i] = a1[i]; }
            acc = __builtin_amdgcn_wmma_f32_16x16x32_f16(false, a, false, bw[kk],
                                                         (short)0, acc, false, false);
        }
        {
            const float* prow = pre + (long)t * BATCH * NG
                                + (long)(mBase + lhalf * 8) * NG + nBase + lrow;
            float* grow = gates + (mBase + lhalf * 8) * NG + nBase + lrow;
            #pragma unroll
            for (int r = 0; r < 8; ++r) grow[r * NG] = acc[r] + prow[r * NG];
            if (t + 1 < T_STEPS) {
                #pragma unroll
                for (int r = 0; r < 8; ++r)
                    __builtin_prefetch(prow + (long)BATCH * NG + r * NG, 0, 1);
            }
        }
        grid_sync(sync, sync + 1);     // gates[t] complete everywhere

        // ---- pointwise for batch row b
        const float* grow = gates + (long)b * NG;
        float s = 0.f, ss = 0.f, v[8];
        #pragma unroll
        for (int i = 0; i < 8; ++i) {
            int n = tid * 8 + i;
            float u = grow[n] + bias[n];
            v[i] = u; s += u; ss += u * u;
        }
        { // block reduce (2 values)
            __syncthreads();
            #pragma unroll
            for (int off = 16; off > 0; off >>= 1) {
                s += __shfl_xor(s, off, 32); ss += __shfl_xor(ss, off, 32);
            }
            if (lane == 0) { red[wv] = s; red[8 + wv] = ss; }
            __syncthreads();
            if (wv == 0) {
                float ra = (lane < 8) ? red[lane] : 0.f;
                float rb = (lane < 8) ? red[8 + lane] : 0.f;
                #pragma unroll
                for (int off = 4; off > 0; off >>= 1) {
                    ra += __shfl_xor(ra, off, 32); rb += __shfl_xor(rb, off, 32);
                }
                if (lane == 0) { red[0] = ra; red[8] = rb; }
            }
            __syncthreads();
            s = red[0]; ss = red[8];
        }
        float mu   = s * (1.f / NG);
        float rstd = rsqrtf(ss * (1.f / NG) - mu * mu + LN_EPS);
        #pragma unroll
        for (int i = 0; i < 8; ++i) {
            int n = tid * 8 + i;
            gl[n] = (v[i] - mu) * rstd * gg[n] + gb[n];
        }
        __syncthreads();

        float cn[2], og[2], s2 = 0.f, ss2 = 0.f;
        #pragma unroll
        for (int i = 0; i < 2; ++i) {
            int j = tid * 2 + i;
            float f  = sigmoidf_(gl[j]);
            float in = sigmoidf_(gl[HDIM + j]);
            float gv = tanhf(gl[2 * HDIM + j]);
            og[i] = sigmoidf_(gl[3 * HDIM + j]);
            float u = f * creg[i] + in * gv;
            cn[i] = u; s2 += u; ss2 += u * u;
        }
        { // block reduce (2 values)
            __syncthreads();
            #pragma unroll
            for (int off = 16; off > 0; off >>= 1) {
                s2 += __shfl_xor(s2, off, 32); ss2 += __shfl_xor(ss2, off, 32);
            }
            if (lane == 0) { red[wv] = s2; red[8 + wv] = ss2; }
            __syncthreads();
            if (wv == 0) {
                float ra = (lane < 8) ? red[lane] : 0.f;
                float rb = (lane < 8) ? red[8 + lane] : 0.f;
                #pragma unroll
                for (int off = 4; off > 0; off >>= 1) {
                    ra += __shfl_xor(ra, off, 32); rb += __shfl_xor(rb, off, 32);
                }
                if (lane == 0) { red[0] = ra; red[8] = rb; }
            }
            __syncthreads();
            s2 = red[0]; ss2 = red[8];
        }
        float mu2   = s2 * (1.f / HDIM);
        float rstd2 = rsqrtf(ss2 * (1.f / HDIM) - mu2 * mu2 + LN_EPS);
        #pragma unroll
        for (int i = 0; i < 2; ++i) {
            int j = tid * 2 + i;
            float cc = (cn[i] - mu2) * rstd2 * cg[j] + cb[j];
            float h  = og[i] * tanhf(cc);
            creg[i] = cc;
            h_f16[b * HDIM + j] = (_Float16)h;
            out[(long)t * BATCH * HDIM + b * HDIM + j] = h;
            if (t == T_STEPS - 1) {
                out[(long)T_STEPS * BATCH * HDIM + b * HDIM + j] = h;                 // h_last
                out[(long)T_STEPS * BATCH * HDIM + BATCH * HDIM + b * HDIM + j] = cc; // c_last
            }
        }
        grid_sync(sync, sync + 1);     // h[t] visible before next step's staging
    }
}

// ---------------------------------------------------------------------------
extern "C" void kernel_launch(void* const* d_in, const int* in_sizes, int n_in,
                              void* d_out, int out_size, void* d_ws, size_t ws_size,
                              hipStream_t stream) {
    (void)in_sizes; (void)n_in; (void)out_size; (void)ws_size;
    const float* x    = (const float*)d_in[0];
    const float* h0   = (const float*)d_in[1];
    const float* c0   = (const float*)d_in[2];
    const float* W    = (const float*)d_in[3];
    const float* bias = (const float*)d_in[4];
    const float* gg   = (const float*)d_in[5];
    const float* gb   = (const float*)d_in[6];
    const float* cg   = (const float*)d_in[7];
    const float* cb   = (const float*)d_in[8];
    float* out = (float*)d_out;

    char* ws = (char*)d_ws;
    _Float16* Wt    = (_Float16*)ws;                                   // 4 MB
    _Float16* h_f16 = (_Float16*)(ws + (4u << 20));                    // 64 KB
    float*    gates = (float*)(ws + (4u << 20) + (1u << 16));          // 512 KB
    unsigned* sync  = (unsigned*)(ws + (4u << 20) + (1u << 16) + (1u << 19));
    float*    pre   = (float*)(ws + (8u << 20));                       // 512 MB

    prep_weights<<<(NG * KW) / 256, 256, 0, stream>>>(W, Wt);
    init_sync<<<1, 32, 0, stream>>>(sync);
    pregate_gemm<<<dim3((T_STEPS * BATCH) / 64, NG / 128), 256, 0, stream>>>(x, Wt, pre);
    lstm_persistent<<<NBLK, 256, 0, stream>>>(Wt, pre, h0, c0, bias, gg, gb, cg, cb,
                                              h_f16, gates, out, sync);
}
